// TransformerBlock_59562606461173
// MI455X (gfx1250) — matrix operand order repaired and checked by reading, not compiled, not run on hardware
//
#include <hip/hip_runtime.h>
#include <math.h>

// Problem dims (fixed by reference setup_inputs)
#define Bz  2
#define Tz  2048
#define Cz  1024
#define Hz  16
#define DHz 64
#define FFz 4096
#define Mz  (Bz*Tz)          // 4096 rows
#define THRESH 0.05f
#define LN_EPS 1e-5f

typedef _Float16 h8  __attribute__((ext_vector_type(8)));
typedef _Float16 h16 __attribute__((ext_vector_type(16)));
typedef float    f8  __attribute__((ext_vector_type(8)));

enum { EPI_F32 = 0, EPI_F16 = 1, EPI_GELU_F16 = 2, EPI_VT_F16 = 3 };

// ---------------------------------------------------------------------------
// WMMA GEMM: Y[M,N] = epilogue( alpha * scale * (X[M,K] @ W[N,K]^T) + bias[N] )
// X: f16 row-major, row stride lda. W: f16 row-major [N,K], row stride ldb
// (column n of the K x N B-matrix == contiguous row n of W).
// One wave -> 16x64 output (4 accumulators). Block = 8 waves arranged WM x WN
// (WM*WN == 8). Grids are launched EXACTLY (no tile guards, no divergence:
// EXEC is all-ones for every WMMA as the ISA requires).
// K must be a multiple of 64 (true for 64 / 1024 / 2048 / 4096).
// ---------------------------------------------------------------------------
template <int WM, int WN, int MODE>
__global__ __launch_bounds__(256)
void gemm_tpl(const _Float16* __restrict__ X, int lda,
              const _Float16* __restrict__ W, int ldb,
              const float* __restrict__ alphap,
              const float* __restrict__ bias,
              void* __restrict__ Yout, int ldo,
              int K, float scale)
{
    const int lane = threadIdx.x & 31;
    const int wid  = threadIdx.x >> 5;
    const int half = lane >> 4;       // lane group 0..15 / 16..31
    const int lm   = lane & 15;       // row (A) / col (B,D) within group

    const int waveM = wid % WM;
    const int waveN = wid / WM;
    const int row0  = (blockIdx.x * WM + waveM) * 16;
    const int col0  = (blockIdx.y * WN + waveN) * 64;

    f8 acc[4];
#pragma unroll
    for (int i = 0; i < 4; ++i)
#pragma unroll
        for (int j = 0; j < 8; ++j) acc[i][j] = 0.0f;

    // A fragment base (contiguous K order, matching B):
    //   lanes 0-15:  v[0..15]->K k0+0..15
    //   lanes16-31:  v[0..15]->K k0+16..31
    const _Float16* aptr = X + (size_t)(row0 + lm) * (size_t)lda + half * 16;
    // B fragment bases: lane col = lm, K = k0 + half*16 + j (16 contiguous)
    const _Float16* bptr[4];
#pragma unroll
    for (int nt = 0; nt < 4; ++nt)
        bptr[nt] = W + (size_t)(col0 + nt * 16 + lm) * (size_t)ldb + half * 16;

    for (int k0 = 0; k0 < K; k0 += 64) {
        // Issue all loads for two K-steps up front so the scheduler can
        // overlap the second stage's loads with the first stage's WMMAs.
        const h8 a00 = *reinterpret_cast<const h8*>(aptr + k0);
        const h8 a01 = *reinterpret_cast<const h8*>(aptr + k0 + 8);
        const h8 a10 = *reinterpret_cast<const h8*>(aptr + k0 + 32);
        const h8 a11 = *reinterpret_cast<const h8*>(aptr + k0 + 40);
        h16 B0[4], B1[4];
#pragma unroll
        for (int nt = 0; nt < 4; ++nt) {
            B0[nt] = *reinterpret_cast<const h16*>(bptr[nt] + k0);
            B1[nt] = *reinterpret_cast<const h16*>(bptr[nt] + k0 + 32);
        }
        const h16 A0 = __builtin_shufflevector(a00, a01, 0, 1, 2, 3, 4, 5, 6, 7,
                                               8, 9, 10, 11, 12, 13, 14, 15);
        const h16 A1 = __builtin_shufflevector(a10, a11, 0, 1, 2, 3, 4, 5, 6, 7,
                                               8, 9, 10, 11, 12, 13, 14, 15);
#pragma unroll
        for (int nt = 0; nt < 4; ++nt)
            acc[nt] = __builtin_amdgcn_wmma_f32_16x16x32_f16(
                false, A0, false, B0[nt], (short)0, acc[nt], false, false);
#pragma unroll
        for (int nt = 0; nt < 4; ++nt)
            acc[nt] = __builtin_amdgcn_wmma_f32_16x16x32_f16(
                false, A1, false, B1[nt], (short)0, acc[nt], false, false);
    }

    const float alpha = alphap ? *alphap : 1.0f;
#pragma unroll
    for (int nt = 0; nt < 4; ++nt) {
        const int col = col0 + nt * 16 + lm;
        const float bv = bias ? bias[col] : 0.0f;
#pragma unroll
        for (int r = 0; r < 8; ++r) {
            const int row = row0 + r + half * 8;  // D layout: VGPR r -> M = r + 8*half
            float v = acc[nt][r] * alpha * scale + bv;
            if constexpr (MODE == EPI_F32) {
                ((float*)Yout)[(size_t)row * ldo + col] = v;
            } else if constexpr (MODE == EPI_F16) {
                ((_Float16*)Yout)[(size_t)row * ldo + col] = (_Float16)v;
            } else if constexpr (MODE == EPI_GELU_F16) {
                float g = 0.5f * v * (1.0f + erff(v * 0.70710678118654752f));
                ((_Float16*)Yout)[(size_t)row * ldo + col] = (_Float16)g;
            } else { // EPI_VT_F16: scatter V into [B,H,DH,T]
                const int b = row >> 11, t = row & (Tz - 1);
                const int h = col >> 6,  d = col & (DHz - 1);
                const size_t idx = ((((size_t)b * Hz + h) * DHz + d) * Tz) + t;
                ((_Float16*)Yout)[idx] = (_Float16)v;
            }
        }
    }
}

// ---------------------------------------------------------------------------
// LayerNorm row kernel: x[row, 0..C) f32 -> f16
// ---------------------------------------------------------------------------
__global__ __launch_bounds__(256)
void layernorm_to_f16(const float* __restrict__ x, const float* __restrict__ g,
                      const float* __restrict__ b, _Float16* __restrict__ out)
{
    __shared__ float red[256];
    const int t = threadIdx.x;
    const float* xr = x + (size_t)blockIdx.x * Cz;
    float v[4], s = 0.0f;
#pragma unroll
    for (int i = 0; i < 4; ++i) { v[i] = xr[t + i * 256]; s += v[i]; }
    red[t] = s; __syncthreads();
    for (int off = 128; off > 0; off >>= 1) { if (t < off) red[t] += red[t + off]; __syncthreads(); }
    const float mu = red[0] * (1.0f / Cz); __syncthreads();
    float s2 = 0.0f;
#pragma unroll
    for (int i = 0; i < 4; ++i) { float d = v[i] - mu; s2 += d * d; }
    red[t] = s2; __syncthreads();
    for (int off = 128; off > 0; off >>= 1) { if (t < off) red[t] += red[t + off]; __syncthreads(); }
    const float rstd = rsqrtf(red[0] * (1.0f / Cz) + LN_EPS);
    _Float16* orow = out + (size_t)blockIdx.x * Cz;
#pragma unroll
    for (int i = 0; i < 4; ++i) {
        const int c = t + i * 256;
        orow[c] = (_Float16)((v[i] - mu) * rstd * g[c] + b[c]);
    }
}

// ---------------------------------------------------------------------------
// Softmax over rows of S[Tz,Tz] f32 -> P f16
// ---------------------------------------------------------------------------
__global__ __launch_bounds__(256)
void softmax_rows(const float* __restrict__ S, _Float16* __restrict__ P)
{
    __shared__ float red[256];
    const int t = threadIdx.x;
    const float* sr = S + (size_t)blockIdx.x * Tz;
    float v[8], mx = -1e30f;
#pragma unroll
    for (int i = 0; i < 8; ++i) { v[i] = sr[t + i * 256]; mx = fmaxf(mx, v[i]); }
    red[t] = mx; __syncthreads();
    for (int off = 128; off > 0; off >>= 1) { if (t < off) red[t] = fmaxf(red[t], red[t + off]); __syncthreads(); }
    const float m = red[0]; __syncthreads();
    float sum = 0.0f;
#pragma unroll
    for (int i = 0; i < 8; ++i) { v[i] = expf(v[i] - m); sum += v[i]; }
    red[t] = sum; __syncthreads();
    for (int off = 128; off > 0; off >>= 1) { if (t < off) red[t] += red[t + off]; __syncthreads(); }
    const float inv = 1.0f / red[0];
    _Float16* pr = P + (size_t)blockIdx.x * Tz;
#pragma unroll
    for (int i = 0; i < 8; ++i) pr[t + i * 256] = (_Float16)(v[i] * inv);
}

// ---------------------------------------------------------------------------
// Reductions / quantizers
// ---------------------------------------------------------------------------
__global__ void zero_scalars(float* s, int n)
{ int i = blockIdx.x * blockDim.x + threadIdx.x; if (i < n) s[i] = 0.0f; }

__global__ __launch_bounds__(256)
void absmax_f32(const float* __restrict__ w, long long n, unsigned* __restrict__ outbits)
{
    __shared__ float red[256];
    const int t = threadIdx.x;
    float m = 0.0f;
    for (long long i = (long long)blockIdx.x * 256 + t; i < n; i += (long long)gridDim.x * 256)
        m = fmaxf(m, fabsf(w[i]));
    red[t] = m; __syncthreads();
    for (int off = 128; off > 0; off >>= 1) { if (t < off) red[t] = fmaxf(red[t], red[t + off]); __syncthreads(); }
    if (t == 0) atomicMax(outbits, __float_as_uint(red[0]));   // vals >= 0: uint order == float order
}

// 512 blocks -> deterministic per-block partials: part[0..511]=sum(|w|*mask), part[512..1023]=sum(mask)
__global__ __launch_bounds__(256)
void tern_sums(const float* __restrict__ w, long long n,
               const unsigned* __restrict__ maxbits, float* __restrict__ part)
{
    __shared__ float r0[256], r1[256];
    const float delta = THRESH * __uint_as_float(*maxbits);
    const int t = threadIdx.x;
    float saw = 0.0f, sm = 0.0f;
    for (long long i = (long long)blockIdx.x * 256 + t; i < n; i += (long long)gridDim.x * 256) {
        const float a = fabsf(w[i]);
        if (a > delta) { saw += a; sm += 1.0f; }
    }
    r0[t] = saw; r1[t] = sm; __syncthreads();
    for (int off = 128; off > 0; off >>= 1) {
        if (t < off) { r0[t] += r0[t + off]; r1[t] += r1[t + off]; }
        __syncthreads();
    }
    if (t == 0) { part[blockIdx.x] = r0[0]; part[512 + blockIdx.x] = r1[0]; }
}

__global__ __launch_bounds__(256)
void finalize_alpha(const float* __restrict__ part, float* __restrict__ alpha)
{
    __shared__ float r0[256], r1[256];
    const int t = threadIdx.x;
    r0[t] = part[t] + part[t + 256];
    r1[t] = part[512 + t] + part[512 + t + 256];
    __syncthreads();
    for (int off = 128; off > 0; off >>= 1) {
        if (t < off) { r0[t] += r0[t + off]; r1[t] += r1[t + off]; }
        __syncthreads();
    }
    if (t == 0) *alpha = r0[0] / fmaxf(r1[0], 1.0f);
}

__global__ __launch_bounds__(256)
void tern_quant(const float* __restrict__ w, long long n,
                const unsigned* __restrict__ maxbits, _Float16* __restrict__ wq)
{
    const float delta = THRESH * __uint_as_float(*maxbits);
    const long long i = (long long)blockIdx.x * 256 + threadIdx.x;
    if (i < n) {
        const float x = w[i];
        const float q = (fabsf(x) > delta) ? (x > 0.0f ? 1.0f : -1.0f) : 0.0f;
        wq[i] = (_Float16)q;
    }
}

__global__ __launch_bounds__(256)
void fp4_quant_inplace(float* __restrict__ y, long long n, const unsigned* __restrict__ maxbits)
{
    const float s = fmaxf(__uint_as_float(*maxbits) * (1.0f / 7.0f), 1e-8f);
    const float inv = 1.0f / s;
    const long long i = (long long)blockIdx.x * 256 + threadIdx.x;
    if (i < n) {
        float q = rintf(y[i] * inv);            // RNE == jnp.round
        q = fminf(fmaxf(q, -7.0f), 7.0f);
        y[i] = q * s;
    }
}

__global__ __launch_bounds__(256)
void add_f32(const float* __restrict__ a, const float* __restrict__ b,
             float* __restrict__ c, long long n)
{
    const long long i = (long long)blockIdx.x * 256 + threadIdx.x;
    if (i < n) c[i] = a[i] + b[i];
}

// ---------------------------------------------------------------------------
extern "C" void kernel_launch(void* const* d_in, const int* in_sizes, int n_in,
                              void* d_out, int out_size, void* d_ws, size_t ws_size,
                              hipStream_t stream)
{
    const float* x   = (const float*)d_in[0];
    const float* wqm = (const float*)d_in[1];  const float* bq = (const float*)d_in[2];
    const float* wkm = (const float*)d_in[3];  const float* bk = (const float*)d_in[4];
    const float* wvm = (const float*)d_in[5];  const float* bv = (const float*)d_in[6];
    const float* wom = (const float*)d_in[7];  const float* bo = (const float*)d_in[8];
    const float* w1m = (const float*)d_in[9];  const float* b1 = (const float*)d_in[10];
    const float* w2m = (const float*)d_in[11]; const float* b2 = (const float*)d_in[12];
    const float* g1  = (const float*)d_in[13]; const float* be1 = (const float*)d_in[14];
    const float* g2  = (const float*)d_in[15]; const float* be2 = (const float*)d_in[16];

    float* out  = (float*)d_out;
    float* xout = out;                       // (B,T,C)
    float* h1   = out + (size_t)Mz * Cz;     // (B,T,C)
    float* h2   = out + 2 * (size_t)Mz * Cz; // (B,T,C)

    // ---- carve workspace ----
    char* wsb = (char*)d_ws;
    size_t off = 0;
    auto carve = [&](size_t bytes) -> char* {
        char* p = wsb + off; off += (bytes + 255) & ~(size_t)255; return p;
    };
    float*     scal  = (float*)carve(64 * sizeof(float));  // [4i..4i+3]: maxbits,-,-,alpha; [24],[25]: fp4 maxes
    float*     part  = (float*)carve(1024 * sizeof(float));
    _Float16*  wqq   = (_Float16*)carve((size_t)Cz * Cz * 2);
    _Float16*  wkq   = (_Float16*)carve((size_t)Cz * Cz * 2);
    _Float16*  wvq   = (_Float16*)carve((size_t)Cz * Cz * 2);
    _Float16*  woq   = (_Float16*)carve((size_t)Cz * Cz * 2);
    _Float16*  w1q   = (_Float16*)carve((size_t)FFz * Cz * 2);
    _Float16*  w2q   = (_Float16*)carve((size_t)Cz * FFz * 2);
    _Float16*  xln1  = (_Float16*)carve((size_t)Mz * Cz * 2);
    _Float16*  qf    = (_Float16*)carve((size_t)Mz * Cz * 2);
    _Float16*  kf    = (_Float16*)carve((size_t)Mz * Cz * 2);
    _Float16*  vT    = (_Float16*)carve((size_t)Mz * Cz * 2);   // [B,H,DH,T]
    _Float16*  yattn = (_Float16*)carve((size_t)Mz * Cz * 2);
    float*     sbuf  = (float*)carve((size_t)Tz * Tz * 4);
    _Float16*  pbuf  = (_Float16*)carve((size_t)Tz * Tz * 2);
    float*     x1    = (float*)carve((size_t)Mz * Cz * 4);
    _Float16*  xln2  = (_Float16*)carve((size_t)Mz * Cz * 2);
    _Float16*  hg    = (_Float16*)carve((size_t)Mz * FFz * 2);

    const long long nC  = (long long)Mz * Cz;      // 4M
    const dim3 blk(256);

    // ---- 0. zero scalar slots (re-done every call -> replay safe) ----
    zero_scalars<<<dim3(1), dim3(64), 0, stream>>>(scal, 64);

    // ---- 1. ternary-quantize all six weights ----
    struct WI { const float* w; long long n; _Float16* q; int slot; };
    const WI wlist[6] = {
        { wqm, (long long)Cz * Cz,  wqq, 0 }, { wkm, (long long)Cz * Cz,  wkq, 1 },
        { wvm, (long long)Cz * Cz,  wvq, 2 }, { wom, (long long)Cz * Cz,  woq, 3 },
        { w1m, (long long)FFz * Cz, w1q, 4 }, { w2m, (long long)Cz * FFz, w2q, 5 },
    };
    for (int i = 0; i < 6; ++i) {
        unsigned* mb = (unsigned*)&scal[4 * wlist[i].slot];
        float*    al = &scal[4 * wlist[i].slot + 3];
        absmax_f32<<<dim3(512), blk, 0, stream>>>(wlist[i].w, wlist[i].n, mb);
        tern_sums <<<dim3(512), blk, 0, stream>>>(wlist[i].w, wlist[i].n, mb, part);
        finalize_alpha<<<dim3(1), blk, 0, stream>>>(part, al);
        tern_quant<<<dim3((unsigned)((wlist[i].n + 255) / 256)), blk, 0, stream>>>(
            wlist[i].w, wlist[i].n, mb, wlist[i].q);
    }
    const float* aQ = &scal[3];  const float* aK = &scal[7];
    const float* aV = &scal[11]; const float* aO = &scal[15];
    const float* a1 = &scal[19]; const float* a2 = &scal[23];

    // ---- 2. LN1 + Q/K/V projections (M=4096, N=1024, K=1024) ----
    layernorm_to_f16<<<dim3(Mz), blk, 0, stream>>>(x, g1, be1, xln1);
    gemm_tpl<4, 2, EPI_F16><<<dim3(Mz / 64, Cz / 128), blk, 0, stream>>>(
        xln1, Cz, wqq, Cz, aQ, bq, qf, Cz, Cz, 1.0f);
    gemm_tpl<4, 2, EPI_F16><<<dim3(Mz / 64, Cz / 128), blk, 0, stream>>>(
        xln1, Cz, wkq, Cz, aK, bk, kf, Cz, Cz, 1.0f);
    gemm_tpl<4, 2, EPI_VT_F16><<<dim3(Mz / 64, Cz / 128), blk, 0, stream>>>(
        xln1, Cz, wvq, Cz, aV, bv, vT, 0, Cz, 1.0f);

    // ---- 3. attention, one (b,h) at a time (reuses 24MB score/prob scratch) ----
    for (int b = 0; b < Bz; ++b) {
        for (int h = 0; h < Hz; ++h) {
            const _Float16* qs = qf + (size_t)b * Tz * Cz + h * DHz;
            const _Float16* ks = kf + (size_t)b * Tz * Cz + h * DHz;
            // scores: M=2048, N=2048, K=64
            gemm_tpl<4, 2, EPI_F32><<<dim3(Tz / 64, Tz / 128), blk, 0, stream>>>(
                qs, Cz, ks, Cz, nullptr, nullptr, sbuf, Tz, DHz, 0.125f);
            softmax_rows<<<dim3(Tz), blk, 0, stream>>>(sbuf, pbuf);
            // P @ V: M=2048, N=64, K=2048  -> 8x1 wave arrangement (block 128x64)
            const _Float16* vs = vT + ((size_t)b * Hz + h) * DHz * Tz;  // [DH,T] row-major
            _Float16* yo = yattn + (size_t)b * Tz * Cz + h * DHz;
            gemm_tpl<8, 1, EPI_F16><<<dim3(Tz / 128, 1), blk, 0, stream>>>(
                pbuf, Tz, vs, Tz, nullptr, nullptr, yo, Cz, Tz, 1.0f);
        }
    }

    // ---- 4. O-projection -> fp4 fake-quant = h1 ----
    gemm_tpl<4, 2, EPI_F32><<<dim3(Mz / 64, Cz / 128), blk, 0, stream>>>(
        yattn, Cz, woq, Cz, aO, bo, h1, Cz, Cz, 1.0f);
    absmax_f32<<<dim3(512), blk, 0, stream>>>(h1, nC, (unsigned*)&scal[24]);
    fp4_quant_inplace<<<dim3((unsigned)((nC + 255) / 256)), blk, 0, stream>>>(
        h1, nC, (unsigned*)&scal[24]);

    // ---- 5. residual, LN2, FFN ----
    add_f32<<<dim3((unsigned)((nC + 255) / 256)), blk, 0, stream>>>(x, h1, x1, nC);
    layernorm_to_f16<<<dim3(Mz), blk, 0, stream>>>(x1, g2, be2, xln2);
    gemm_tpl<4, 2, EPI_GELU_F16><<<dim3(Mz / 64, FFz / 128), blk, 0, stream>>>(
        xln2, Cz, w1q, Cz, a1, b1, hg, FFz, Cz, 1.0f);
    gemm_tpl<4, 2, EPI_F32><<<dim3(Mz / 64, Cz / 128), blk, 0, stream>>>(
        hg, FFz, w2q, FFz, a2, b2, h2, Cz, FFz, 1.0f);
    absmax_f32<<<dim3(512), blk, 0, stream>>>(h2, nC, (unsigned*)&scal[25]);
    fp4_quant_inplace<<<dim3((unsigned)((nC + 255) / 256)), blk, 0, stream>>>(
        h2, nC, (unsigned*)&scal[25]);

    // ---- 6. final residual ----
    add_f32<<<dim3((unsigned)((nC + 255) / 256)), blk, 0, stream>>>(x1, h2, xout, nC);
}
